// vMF_27779848470877
// MI455X (gfx1250) — matrix operand order repaired and checked by reading, not compiled, not run on hardware
//
#include <hip/hip_runtime.h>
#include <hip/hip_bf16.h>

// ---- vector types for WMMA (probe-verified signatures) ----
typedef _Float16 v16h __attribute__((ext_vector_type(16)));
typedef _Float16 v8h  __attribute__((ext_vector_type(8)));
typedef _Float16 v4h  __attribute__((ext_vector_type(4)));
typedef float    v8f  __attribute__((ext_vector_type(8)));
typedef float    v4f  __attribute__((ext_vector_type(4)));

union HFrag { v16h v; v8h h[2]; };

#define KAPPA   50.0f
#define DIM     512
#define DM1     511.0f      // dim - 1
#define NORMEPS 1.0f
#define NORMMAX 9.0f

// ---------------- stateless RNG helpers ----------------
static __device__ __forceinline__ unsigned hashu(unsigned x) {
    x ^= x >> 16; x *= 0x7feb352du;
    x ^= x >> 15; x *= 0x846ca68bu;
    x ^= x >> 16; return x;
}
static __device__ __forceinline__ float u01(unsigned h) {
    return (float)(h >> 8) * 5.9604645e-08f + 2.9802322e-08f; // (0,1)
}
// Accurate Box-Muller normal (used only in the 64-thread/WG row sampler).
static __device__ __forceinline__ float gaussian(unsigned e, unsigned seed) {
    unsigned h1 = hashu(e ^ seed);
    unsigned h2 = hashu(e + 0x9e3779b9u + (seed << 1));
    float r = sqrtf(fmaxf(-2.0f * __logf(u01(h1)), 0.0f));
    return r * __cosf(6.28318531f * u01(h2));
}
// Cheap per-element normal: popcount of a hash -> Binomial(32,1/2) ~ N(16, 8).
// ~9 VALU ops, no transcendentals; keeps elementwise noise cost under the
// HBM-roofline time so it co-executes with the WMMA pipe.
static __device__ __forceinline__ float gauss_fast(unsigned e, unsigned seed) {
    unsigned h = hashu(e ^ seed);
    return (float)(__popc(h) - 16) * 0.35355339f;  // 1/sqrt(8)
}

// Wood (1994) rejection sampler for vMF radial weight (kappa=50, d=511).
// Beta(255.5,255.5) drawn via its normal approximation (sd = 0.5/sqrt(2a+1)).
static __device__ __forceinline__ float sample_w(unsigned row, unsigned seed) {
    const float bb = 0.823305f;     // d / (sqrt(4k^2+d^2) + 2k)
    const float xx = 0.0969105f;    // (1-b)/(1+b)
    const float cc = KAPPA * xx + DM1 * __logf(1.0f - xx * xx);
    float w = 0.0f;
    bool done = false;
#pragma unroll 4
    for (int i = 0; i < 16; ++i) {
        float n = gaussian(row * 37u + (unsigned)i, seed ^ 0x85ebca6bu);
        float z = 0.5f + 0.0220971f * n;
        z = fminf(fmaxf(z, 1e-6f), 1.0f - 1e-6f);
        float wc = (1.0f - (1.0f + bb) * z) / (1.0f - (1.0f - bb) * z);
        float u  = u01(hashu(row * 41u + 0x51edu + (unsigned)i + seed));
        bool ok = (KAPPA * wc + DM1 * __logf(fmaxf(1.0f - xx * wc, 1e-20f)) - cc) >= __logf(u);
        if (ok && !done) { w = wc; done = true; }
    }
    return w;
}

// ---------------- fused GEMM + vMF sampling kernel ----------------
// Grid: B/64 workgroups of 512 threads (16 waves).
// Each WG: rows [row0, row0+64) x all 512 columns.
// Wave w: M-tile (w&3), N-tiles (w>>2)*8 .. +7  -> 8 16x16 f32 accumulators.
// K pipeline: double-buffered LDS (74 KB; CDNA5 WGP has 320 KB), one barrier
// per K-step; global loads for step s+1 issued before the WMMAs of step s.
__global__ __launch_bounds__(512, 1) void vmf_fused_kernel(
    const float* __restrict__ latent, const float* __restrict__ W,
    const float* __restrict__ bias, float* __restrict__ out,
    int B, unsigned sseed)
{
    __shared__ alignas(16) _Float16 sA[2][64 * 32];    // latent tile [row][k]
    __shared__ alignas(16) _Float16 sB[2][512 * 32];   // W tile      [col][k]
    __shared__ float sMu2[64], sMuV0[64], sV02[64], sRd2[64];
    __shared__ float sC1[64], sC2[64], sC3[64];

    const int tid    = threadIdx.x;
    const int wave   = tid >> 5;
    const int lane   = tid & 31;
    const int hl     = lane & 15;            // lane within half-wave
    const int hiHalf = (lane >= 16) ? 1 : 0;
    const int mtile  = wave & 3;             // 0..3
    const int ngrp   = wave >> 2;            // 0..3  (8 N-tiles each)
    const int row0   = blockIdx.x * 64;

    const unsigned SEED_V0 = sseed ^ 0x3779c1f3u;
    const unsigned SEED_RD = sseed ^ 0xc2b2ae35u;

    v8f acc[8];
#pragma unroll
    for (int t = 0; t < 8; ++t) acc[t] = (v8f)0.0f;

    // staging addresses (fixed across K-steps)
    const int ar  = tid >> 3;                 // A tile row this thread stages
    const int akq = (tid & 7) * 4;            // A tile k-quad
    const int arr = min(row0 + ar, B - 1);
    const float* aptr = latent + (size_t)arr * DIM + akq;
    const float* wptr = W + (size_t)tid * DIM;      // this thread stages W col 'tid'

    v4f aReg;            // latent staging (4 floats)
    v4f wReg[8];         // W staging (32 floats)

    auto loadStep = [&](int k0) {
        aReg = *(const v4f*)(aptr + k0);
#pragma unroll
        for (int q = 0; q < 8; ++q) wReg[q] = *(const v4f*)(wptr + k0 + q * 4);
    };
    auto stageStep = [&](int buf) {
        v4h h4 = { (_Float16)aReg.x, (_Float16)aReg.y, (_Float16)aReg.z, (_Float16)aReg.w };
        *(v4h*)(&sA[buf][ar * 32 + akq]) = h4;
        _Float16* dp = &sB[buf][tid * 32];
#pragma unroll
        for (int q = 0; q < 8; ++q) {
            v4h w4 = { (_Float16)wReg[q].x, (_Float16)wReg[q].y,
                       (_Float16)wReg[q].z, (_Float16)wReg[q].w };
            *(v4h*)(dp + q * 4) = w4;
        }
    };

    // ---------------- pipelined K loop ----------------
    loadStep(0);
    stageStep(0);
    __syncthreads();

    const int NSTEP = DIM / 32;   // 16
    for (int s = 0; s < NSTEP; ++s) {
        const int cur = s & 1;

        // issue global loads for the next tile before computing this one
        if (s + 1 < NSTEP) {
            loadStep((s + 1) * 32);
            if (s + 2 < NSTEP)   // prefetch hint two steps ahead
                __builtin_prefetch(aptr + (s + 2) * 32, 0, 3);
        }

        // A fragment: 16x32 f16. lanes 0-15: K{0..7,16..23}; lanes 16-31: K{8..15,24..31}
        HFrag af;
        {
            const int arow = mtile * 16 + hl;
            const int ak   = hiHalf ? 8 : 0;
            af.h[0] = *(const v8h*)(&sA[cur][arow * 32 + ak]);
            af.h[1] = *(const v8h*)(&sA[cur][arow * 32 + ak + 16]);
        }
        // B fragments in groups of 4, then 4 back-to-back WMMAs
#pragma unroll
        for (int gset = 0; gset < 2; ++gset) {
            HFrag bf[4];
#pragma unroll
            for (int j = 0; j < 4; ++j) {
                const int t  = gset * 4 + j;
                const int bj = (ngrp * 8 + t) * 16 + hl;
                const int bk = hiHalf ? 8 : 0;
                bf[j].h[0] = *(const v8h*)(&sB[cur][bj * 32 + bk]);
                bf[j].h[1] = *(const v8h*)(&sB[cur][bj * 32 + bk + 16]);
            }
#pragma unroll
            for (int j = 0; j < 4; ++j) {
                acc[gset * 4 + j] = __builtin_amdgcn_wmma_f32_16x16x32_f16(
                    false, af.v, false, bf[j].v, (short)0, acc[gset * 4 + j], false, false);
            }
        }

        // convert + stage next tile into the other buffer, then one barrier
        if (s + 1 < NSTEP) stageStep(cur ^ 1);
        __syncthreads();
    }

    // ---------------- bias add ----------------
#pragma unroll
    for (int t = 0; t < 8; ++t) {
        float bv = bias[(ngrp * 8 + t) * 16 + hl];
#pragma unroll
        for (int g = 0; g < 8; ++g) acc[t][g] += bv;
    }

    // ---------------- per-row statistics ----------------
    if (tid < 64) { sMu2[tid] = 0.f; sMuV0[tid] = 0.f; sV02[tid] = 0.f; sRd2[tid] = 0.f; }
    __syncthreads();

#pragma unroll
    for (int g = 0; g < 8; ++g) {
        int rl = mtile * 16 + g + hiHalf * 8;
        unsigned grow = (unsigned)(row0 + rl);
        float s1 = 0.f, s2 = 0.f, s3 = 0.f, s4 = 0.f;
#pragma unroll
        for (int t = 0; t < 8; ++t) {
            unsigned col = (unsigned)((ngrp * 8 + t) * 16 + hl);
            unsigned e = grow * (unsigned)DIM + col;
            float mu = acc[t][g];
            float v0 = gauss_fast(e, SEED_V0);
            float rd = gauss_fast(e, SEED_RD);
            s1 += mu * mu; s2 += mu * v0; s4 += v0 * v0; s3 += rd * rd;
        }
        // reduce the 16 lanes that hold this row (xor masks stay within half-wave)
#pragma unroll
        for (int m = 1; m < 16; m <<= 1) {
            s1 += __shfl_xor(s1, m); s2 += __shfl_xor(s2, m);
            s3 += __shfl_xor(s3, m); s4 += __shfl_xor(s4, m);
        }
        if (hl == 0) {
            atomicAdd(&sMu2[rl], s1); atomicAdd(&sMuV0[rl], s2);
            atomicAdd(&sRd2[rl], s3); atomicAdd(&sV02[rl], s4);
        }
    }
    __syncthreads();

    // ---------------- per-row sampler -> 3 coefficients ----------------
    if (tid < 64) {
        unsigned grow = (unsigned)(row0 + tid);
        float mu2    = sMu2[tid];
        float munorm = sqrtf(mu2);
        float muv0   = sMuV0[tid];
        float v02    = sV02[tid];
        float rd2    = sRd2[tid];

        float w  = sample_w(grow, sseed);
        float un = u01(hashu(grow ^ (0xa511e9b3u + sseed)));
        float rn = u01(hashu(grow + 0x27d4eb2fu + sseed));

        float munoise = fminf(fmaxf(munorm, 0.0f), NORMMAX) + un * NORMEPS;
        float ortho2  = fmaxf(v02 - muv0 * muv0 / fmaxf(mu2, 1e-20f), 1e-20f);
        float invo    = rsqrtf(ortho2);
        float sqw     = sqrtf(fmaxf(1.0f - w * w, 0.0f));
        bool  ok      = munorm > 1e-10f;

        float c1 = munoise * sqw * invo;                                        // * v0
        float c2 = munoise * (w / fmaxf(munorm, 1e-10f)
                              - sqw * invo * muv0 / fmaxf(mu2, 1e-20f));        // * mu
        float c3 = rn * NORMEPS * rsqrtf(fmaxf(rd2, 1e-20f));                   // * rd

        sC1[tid] = ok ? c1 : 0.0f;
        sC2[tid] = ok ? c2 : 0.0f;
        sC3[tid] = ok ? 0.0f : c3;
    }
    __syncthreads();

    // ---------------- recombine + store ----------------
#pragma unroll
    for (int t = 0; t < 8; ++t) {
#pragma unroll
        for (int g = 0; g < 8; ++g) {
            int rl = mtile * 16 + g + hiHalf * 8;
            int grow = row0 + rl;
            if (grow >= B) continue;
            unsigned col = (unsigned)((ngrp * 8 + t) * 16 + hl);
            unsigned e = (unsigned)grow * (unsigned)DIM + col;
            float mu = acc[t][g];
            float v0 = gauss_fast(e, SEED_V0);
            float rd = gauss_fast(e, SEED_RD);
            out[(size_t)grow * DIM + col] = sC1[rl] * v0 + sC2[rl] * mu + sC3[rl] * rd;
        }
    }
}

extern "C" void kernel_launch(void* const* d_in, const int* in_sizes, int n_in,
                              void* d_out, int out_size, void* d_ws, size_t ws_size,
                              hipStream_t stream) {
    const float* latent = (const float*)d_in[0];
    const float* W      = (const float*)d_in[1];
    const float* bias   = (const float*)d_in[2];
    float* out = (float*)d_out;

    const int D = in_sizes[2];                  // 512
    const int B = in_sizes[0] / (D > 0 ? D : 1);
    int ns = (B > 0 && D > 0) ? out_size / (B * D) : 1;
    if (ns < 1) ns = 1;

    const int grid = (B + 63) / 64;
    for (int s = 0; s < ns; ++s) {
        unsigned sseed = 0x9e3779b9u * (unsigned)(s + 1);
        vmf_fused_kernel<<<grid, 512, 0, stream>>>(
            latent, W, bias, out + (size_t)s * B * D, B, sseed);
    }
}
